// Encoder_25340307046697
// MI455X (gfx1250) — matrix operand order repaired and checked by reading, not compiled, not run on hardware
//
#include <hip/hip_runtime.h>
#include <stdint.h>

typedef __attribute__((ext_vector_type(16))) _Float16   v16h;
typedef __attribute__((ext_vector_type(8)))  _Float16   v8h;
typedef __attribute__((ext_vector_type(8)))  float      v8f;
typedef __attribute__((ext_vector_type(8)))  unsigned   v8u;
typedef __attribute__((ext_vector_type(4)))  unsigned   u32x4;
typedef __attribute__((ext_vector_type(8)))  int        i32x8;
typedef __attribute__((ext_vector_type(4)))  int        i32x4;

#define NUM_CLASS 5
#define EMBED     256
#define BATCH     2048
#define NNEI      64
#define NMODES    20
#define INDIM     80
#define KPAD      96

// ---------------------------------------------------------------- WMMA core
static __device__ __forceinline__ v8f wmma16(v16h a, v16h b, v8f c) {
  // D = A(16x32 f16) * B(32x16 f16) + C(16x16 f32)
  return __builtin_amdgcn_wmma_f32_16x16x32_f16(false, a, false, b, (short)0, c,
                                                false, false);
}

// Per-row class mask: in the A layout each lane holds exactly one row, so
// masking a row by its routing label is a lane-uniform AND over the fragment.
static __device__ __forceinline__ v16h maskA(v16h a, bool on) {
  v8u u = __builtin_bit_cast(v8u, a);
  u &= (on ? 0xFFFFFFFFu : 0u);
  return __builtin_bit_cast(v16h, u);
}

// ------------------------------------------------------- TDM weight staging
// One 2-D tensor_load_to_lds: nrows rows of row_elems f16 elements,
// contiguous in global and in LDS.  D# per CDNA5 ISA ch.8:
//  group0: [1:0]=count=1, [63:32]=lds_addr, [120:64]=global_addr, [127:126]=2
//  group1: [17:16]=data_size(1=2B), [79:48]=tensor_dim0, [111:80]=tensor_dim1,
//          [127:112]=tile_dim0, [143:128]=tile_dim1, [207:160]=dim0_stride
static __device__ __forceinline__ void tdm_load_2d(unsigned lds_off,
                                                   const void* gptr,
                                                   unsigned row_elems,
                                                   unsigned nrows) {
  unsigned long long ga = (unsigned long long)(uintptr_t)gptr;
  u32x4 g0;
  g0[0] = 1u;                                   // count=1 (user descriptor)
  g0[1] = lds_off;                              // LDS byte address
  g0[2] = (unsigned)(ga & 0xFFFFFFFFu);
  g0[3] = (unsigned)((ga >> 32) & 0x01FFFFFFu) | 0x80000000u;  // type=2
  i32x8 g1;
  g1[0] = (int)(1u << 16);                      // wg_mask=0, data_size=2B
  g1[1] = (int)((row_elems & 0xFFFFu) << 16);   // tensor_dim0[15:0]
  g1[2] = (int)((row_elems >> 16) & 0xFFFFu) | (int)((nrows & 0xFFFFu) << 16);
  g1[3] = (int)((row_elems & 0xFFFFu) << 16);   // tile_dim0
  g1[4] = (int)(nrows & 0xFFFFu);               // tile_dim1 (tile_dim2=0)
  g1[5] = (int)row_elems;                       // tensor_dim0_stride[31:0]
  g1[6] = 0;
  g1[7] = 0;
  i32x4 z4 = {0, 0, 0, 0};
  i32x8 z8 = {0, 0, 0, 0, 0, 0, 0, 0};
  // 6-arg form (clang-23 / therock headers): (g0, g1, g2, g3, g4, cpol)
  __builtin_amdgcn_tensor_load_to_lds(g0, g1, z4, z4, z8, 0);
}

// B fragment (32x16 f16) from LDS: lane col = n0+(lane&15), lanes>=16 take
// K+16; 16 contiguous halfs per lane (32B aligned -> 2x ds_load_b128).
static __device__ __forceinline__ v16h ldsBfrag(const _Float16* base, int col,
                                                int stride, int ks) {
  return *(const v16h*)(base + col * stride + ks);
}

// One 8-element contiguous run of the A fragment, from an 80-float f32 row.
// Runs start at multiples of 8 so the k<80 predicate is uniform per run and half-wave.
template <bool RECIP>
static __device__ __forceinline__ void loadArun(v16h& a, int off,
                                                const float* __restrict__ rowp,
                                                int ks) {
  if (ks < INDIM) {
    float4 lo = *(const float4*)(rowp + ks);
    float4 hi = *(const float4*)(rowp + ks + 4);
    float t[8] = {lo.x, lo.y, lo.z, lo.w, hi.x, hi.y, hi.z, hi.w};
#pragma unroll
    for (int i = 0; i < 8; ++i) {
      float v = t[i];
      if (RECIP) {
        float d = v + ((v >= 0.f) ? 1e-4f : -1e-4f);
        v = __builtin_amdgcn_rcpf(d);
      }
      a[off + i] = (_Float16)v;
    }
  } else {
#pragma unroll
    for (int i = 0; i < 8; ++i) a[off + i] = (_Float16)0.f;
  }
}

// A fragment (16x32 f16): lane<16 -> row=lane, K in {0..15};
// lane>=16 -> row=lane-16, K in {16..31}.
template <bool RECIP>
static __device__ __forceinline__ v16h loadAfrag(const float* __restrict__ rowp,
                                                 int kbase, int h) {
  v16h a;
  loadArun<RECIP>(a, 0, rowp, kbase + h * 16);
  loadArun<RECIP>(a, 8, rowp, kbase + h * 16 + 8);
  return a;
}

// ------------------------------------------------- weight prep (f32 -> f16^T)
__global__ void __launch_bounds__(256, 1)
prep_weights(const float* __restrict__ neiW,
             const float* __restrict__ obsW,
             const float* __restrict__ modeW,
             _Float16* __restrict__ neiWt,
             _Float16* __restrict__ obsWt,
             _Float16* __restrict__ wtopT) {
  const int S0 = 6 * EMBED * KPAD;
  const int S1 = NUM_CLASS * EMBED * KPAD;
  const int S2 = EMBED * EMBED;
  int idx = blockIdx.x * 256 + threadIdx.x;
  if (idx < S0) {
    int c = idx / (EMBED * KPAD);
    int rem = idx - c * (EMBED * KPAD);
    int n = rem / KPAD, k = rem - n * KPAD;
    float v = (k < INDIM) ? neiW[((size_t)c * INDIM + k) * EMBED + n] : 0.f;
    neiWt[idx] = (_Float16)v;
  } else if (idx < S0 + S1) {
    int j = idx - S0;
    int c = j / (EMBED * KPAD);
    int rem = j - c * (EMBED * KPAD);
    int n = rem / KPAD, k = rem - n * KPAD;
    float v = (k < INDIM) ? obsW[((size_t)c * INDIM + k) * EMBED + n] : 0.f;
    obsWt[j] = (_Float16)v;
  } else if (idx < S0 + S1 + S2) {
    int j = idx - S0 - S1;
    int n = j / EMBED, k = j - n * EMBED;
    wtopT[j] = (_Float16)modeW[(size_t)k * EMBED + n];
  }
}

// P[c][m][n] = modes[c,m,:] @ mode_W[256:,:] + mode_b   (tiny: 13 MFLOP)
__global__ void __launch_bounds__(256, 1)
prep_p(const float* __restrict__ modes,
       const float* __restrict__ modeW,
       const float* __restrict__ modeB,
       float* __restrict__ P) {
  int idx = blockIdx.x * 256 + threadIdx.x;  // < 5*20*256
  int n = idx & (EMBED - 1);
  int cm = idx >> 8;
  float acc = modeB[n];
  const float* mrow = modes + (size_t)cm * EMBED;
  const float* wcol = modeW + (size_t)EMBED * EMBED + n;
  for (int k = 0; k < EMBED; ++k) acc += mrow[k] * wcol[(size_t)k * EMBED];
  P[idx] = acc;
}

// ---------------------------------------- routed (one-hot) embedding via WMMA
// Weights for all experts staged to LDS by TDM; D = sum_c maskA_c(A) @ W_c.
// One block per WGP (big LDS), so tell the compiler: min occupancy 1 block ->
// large VGPR budget, no inner-loop spills.
template <int NCLS, bool RECIP, bool OUT_F16>
__global__ void __launch_bounds__(256, 1)
routed_embed(const float* __restrict__ X,
             const int* __restrict__ labels,
             const _Float16* __restrict__ Wt,
             const float* __restrict__ bias,
             float* __restrict__ outF,
             _Float16* __restrict__ outH) {
  extern __shared__ _Float16 smem[];  // NCLS * EMBED * KPAD halfs
  const int tid = threadIdx.x;
  const int wave = tid >> 5, lane = tid & 31;
  const int lanelo = lane & 15, h = lane >> 4;
  const int row0 = blockIdx.x * 128 + wave * 16;

  if (tid < 32)  // wave 0 launches the weight DMA (overlaps A loads below)
    tdm_load_2d((unsigned)(uintptr_t)smem, Wt, EMBED * KPAD, NCLS);

  const int myrow = row0 + lanelo;
  const int mylab = labels[myrow];
  const float* rowp = X + (size_t)myrow * INDIM;
  v16h A0 = loadAfrag<RECIP>(rowp, 0, h);
  v16h A1 = loadAfrag<RECIP>(rowp, 32, h);
  v16h A2 = loadAfrag<RECIP>(rowp, 64, h);
  int labr[8];
#pragma unroll
  for (int r = 0; r < 8; ++r) labr[r] = labels[row0 + r + h * 8];
  const int h16 = h * 16;

  if (tid < 32) __builtin_amdgcn_s_wait_tensorcnt(0);
  __syncthreads();

#pragma unroll
  for (int g = 0; g < 8; ++g) {  // 8 groups of 2 N-tiles
    v8f acc[2];
#pragma unroll
    for (int nt = 0; nt < 2; ++nt)
      acc[nt] = (v8f){0.f, 0.f, 0.f, 0.f, 0.f, 0.f, 0.f, 0.f};
#pragma unroll
    for (int c = 0; c < NCLS; ++c) {
      const bool on = (mylab == c);
      v16h am[3] = {maskA(A0, on), maskA(A1, on), maskA(A2, on)};
      const _Float16* wc = smem + c * EMBED * KPAD;
      v16h Bf[6];  // batch all ds loads, then all WMMAs (partial dscnt waits)
#pragma unroll
      for (int nt = 0; nt < 2; ++nt) {
        const int col = (g * 2 + nt) * 16 + lanelo;
#pragma unroll
        for (int kc = 0; kc < 3; ++kc)
          Bf[nt * 3 + kc] = ldsBfrag(wc, col, KPAD, kc * 32 + h16);
      }
#pragma unroll
      for (int nt = 0; nt < 2; ++nt)
#pragma unroll
        for (int kc = 0; kc < 3; ++kc)
          acc[nt] = wmma16(am[kc], Bf[nt * 3 + kc], acc[nt]);
    }
#pragma unroll
    for (int nt = 0; nt < 2; ++nt) {
      const int N = (g * 2 + nt) * 16 + lanelo;
      float bv[8];
#pragma unroll
      for (int r = 0; r < 8; ++r) bv[r] = bias[labr[r] * EMBED + N];
#pragma unroll
      for (int r = 0; r < 8; ++r) {  // C/D layout: M = r + 8*h, N = lane&15
        const int rowr = row0 + r + h * 8;
        const float v = acc[nt][r] + bv[r];
        if constexpr (OUT_F16)
          outH[(size_t)rowr * EMBED + N] = (_Float16)v;
        else
          outF[(size_t)rowr * EMBED + N] = v;
      }
    }
  }
}

// ------------------- out[b,m,:] = x[b] @ Wtop + P[label[b], m, :]  (WMMA GEMM)
__global__ void __launch_bounds__(256, 1)
mode_out_kernel(const _Float16* __restrict__ xh,
                const int* __restrict__ slab,
                const _Float16* __restrict__ wtopT,
                const float* __restrict__ P,
                float* __restrict__ out) {
  extern __shared__ _Float16 smem[];  // EMBED*EMBED halfs (Wtop^T)
  const int tid = threadIdx.x;
  const int wave = tid >> 5, lane = tid & 31;
  const int lanelo = lane & 15, h = lane >> 4;
  const int row0 = blockIdx.x * 128 + wave * 16;

  if (tid < 32)
    tdm_load_2d((unsigned)(uintptr_t)smem, wtopT, EMBED, EMBED);

  const int myrow = row0 + lanelo;
  const _Float16* rowp = xh + (size_t)myrow * EMBED;
  v16h A[8];
#pragma unroll
  for (int kc = 0; kc < 8; ++kc) {
    v8h lo = *(const v8h*)(rowp + kc * 32 + h * 16);
    v8h hi = *(const v8h*)(rowp + kc * 32 + h * 16 + 8);
    v16h a;
#pragma unroll
    for (int i = 0; i < 8; ++i) { a[i] = lo[i]; a[i + 8] = hi[i]; }
    A[kc] = a;
  }
  int lab[8];
#pragma unroll
  for (int r = 0; r < 8; ++r) lab[r] = slab[row0 + r + h * 8];
  const int h16 = h * 16;

  if (tid < 32) __builtin_amdgcn_s_wait_tensorcnt(0);
  __syncthreads();

  for (int nt = 0; nt < 16; ++nt) {
    const int col = nt * 16 + lanelo;
    v8f acc = (v8f){0.f, 0.f, 0.f, 0.f, 0.f, 0.f, 0.f, 0.f};
    // two batches of 4 B fragments: 8 ds_load_b128 in flight per batch,
    // half the register pressure of a full batch of 8 (avoids spills)
#pragma unroll
    for (int half = 0; half < 2; ++half) {
      v16h Bf[4];
#pragma unroll
      for (int kc = 0; kc < 4; ++kc)
        Bf[kc] = ldsBfrag(smem, col, EMBED, (half * 4 + kc) * 32 + h16);
#pragma unroll
      for (int kc = 0; kc < 4; ++kc)
        acc = wmma16(A[half * 4 + kc], Bf[kc], acc);
    }

    const int N = col;
#pragma unroll
    for (int r = 0; r < 8; ++r) {
      const int rowr = row0 + r + h * 8;
      const float yv = acc[r];
      const float* Pp = P + ((size_t)lab[r] * NMODES) * EMBED + N;
      float* op = out + ((size_t)rowr * NMODES) * EMBED + N;
      float pv[20];
#pragma unroll
      for (int m = 0; m < NMODES; ++m) pv[m] = Pp[(size_t)m * EMBED];
#pragma unroll
      for (int m = 0; m < NMODES; ++m) op[(size_t)m * EMBED] = yv + pv[m];
    }
  }
}

// ---------------------------------------------------------------------- launch
extern "C" void kernel_launch(void* const* d_in, const int* in_sizes, int n_in,
                              void* d_out, int out_size, void* d_ws, size_t ws_size,
                              hipStream_t stream) {
  const float* obs      = (const float*)d_in[0];
  const float* neis     = (const float*)d_in[1];
  const int*   self_lab = (const int*)d_in[2];
  const int*   nei_lab  = (const int*)d_in[3];
  const float* modes    = (const float*)d_in[4];
  const float* obs_W    = (const float*)d_in[5];
  const float* obs_b    = (const float*)d_in[6];
  const float* nei_W    = (const float*)d_in[7];
  const float* nei_b    = (const float*)d_in[8];
  const float* mode_W   = (const float*)d_in[9];
  const float* mode_b   = (const float*)d_in[10];

  float* out    = (float*)d_out;                          // [B,20,256]
  float* neiOut = out + (size_t)BATCH * NMODES * EMBED;   // [B,64,256]

  // workspace layout (all 32B aligned; total ~1.8 MB)
  _Float16* neiWt = (_Float16*)d_ws;                      // 6*256*96 halfs
  _Float16* obsWt = neiWt + 6 * EMBED * KPAD;             // 5*256*96 halfs
  _Float16* wtopT = obsWt + NUM_CLASS * EMBED * KPAD;     // 256*256 halfs
  float*    Ptab  = (float*)(wtopT + EMBED * EMBED);      // 5*20*256 f32
  _Float16* xh    = (_Float16*)(Ptab + NUM_CLASS * NMODES * EMBED);  // 2048*256

  const int SW = 6 * EMBED * KPAD + NUM_CLASS * EMBED * KPAD + EMBED * EMBED;
  prep_weights<<<(SW + 255) / 256, 256, 0, stream>>>(nei_W, obs_W, mode_W,
                                                     neiWt, obsWt, wtopT);
  prep_p<<<(NUM_CLASS * NMODES * EMBED) / 256, 256, 0, stream>>>(modes, mode_W,
                                                                 mode_b, Ptab);
  // x = routed obs embed -> f16 (feeds mode GEMM); 5 experts in 240KB LDS
  routed_embed<NUM_CLASS, false, true>
      <<<BATCH / 128, 256, NUM_CLASS * EMBED * KPAD * sizeof(_Float16), stream>>>(
          obs, self_lab, obsWt, obs_b, (float*)nullptr, xh);
  // nei_feats = routed reciprocal embed (dominant, 134 MB); 6 experts in 288KB LDS
  routed_embed<NUM_CLASS + 1, true, false>
      <<<(BATCH * NNEI) / 128, 256, (NUM_CLASS + 1) * EMBED * KPAD * sizeof(_Float16),
         stream>>>(neis, nei_lab, neiWt, nei_b, neiOut, (_Float16*)nullptr);
  // out = x @ Wtop (Wtop^T staged in 128KB LDS) + P broadcast over 20 modes
  mode_out_kernel<<<BATCH / 128, 256, EMBED * EMBED * sizeof(_Float16), stream>>>(
      xh, self_lab, wtopT, Ptab, out);
}
